// FC_CharNet_9749575762824
// MI455X (gfx1250) — matrix operand order repaired and checked by reading, not compiled, not run on hardware
//
#include <hip/hip_runtime.h>
#include <hip/hip_bf16.h>

typedef __bf16 bf16_t;
typedef __attribute__((ext_vector_type(16))) __bf16 v16bf;
typedef __attribute__((ext_vector_type(8)))  __bf16 v8bf;
typedef __attribute__((ext_vector_type(8)))  float  v8f;

// ---------------------------------------------------------------------------
// Conversion/staging kernels: f32 -> bf16 with K padded to Kp (zero filled).
// Padding => no K tails in the GEMM and 16B-aligned fragment loads everywhere.
// ---------------------------------------------------------------------------
__global__ __launch_bounds__(256) void cvt_pad_bf16(
    const float* __restrict__ src, bf16_t* __restrict__ dst,
    int M, int K, int Kp)
{
  int idx = blockIdx.x * 256 + threadIdx.x;
  if (idx >= M * Kp) return;
  int r = idx / Kp, k = idx - r * Kp;
  float v = (k < K) ? src[(size_t)r * K + k] : 0.0f;
  dst[idx] = (bf16_t)v;
}

// layer-1 inputs: [B=128, P=10, S=8, K] gathered into (p,b,s) row order
__global__ __launch_bounds__(256) void cvt_perm_bf16(
    const float* __restrict__ src, bf16_t* __restrict__ dst, int K, int Kp)
{
  int idx = blockIdx.x * 256 + threadIdx.x;
  if (idx >= 10240 * Kp) return;
  int r = idx / Kp, k = idx - r * Kp;
  int p = r / 1024, rem = r - p * 1024, b = rem >> 3, s = rem & 7;
  float v = (k < K) ? src[(size_t)(((b * 10 + p) << 3) + s) * K + k] : 0.0f;
  dst[idx] = (bf16_t)v;
}

// ---------------------------------------------------------------------------
// GEMM:  Out[M,N] = act( A[M,Kp] @ W[N,Kp]^T + bias ),  all-bf16 WMMA, f32 acc
//   - 256 threads = 8 waves; wave (rowg, colg) computes a 32x64 tile
//     (2 row-tiles x 4 col-tiles -> 8 WMMAs per 32-deep K chunk,
//      A frags reused 4x, B frags 2x; no conversions in the loop).
//   - A and W are pre-padded bf16 with identical padded depth Kp (mult of 32),
//     so the K loop needs no tail and all fragment loads are 16B aligned b128s.
//   - OutF (f32) and/or OutB (bf16, stride N) selected per call.
//   - M multiple of 64 (10240, 1280); bf16 outputs have N multiple of 32.
// ---------------------------------------------------------------------------
__global__ __launch_bounds__(256) void gemm_wmma(
    const bf16_t* __restrict__ A, const bf16_t* __restrict__ W,
    const float* __restrict__ bias,
    float* __restrict__ OutF, bf16_t* __restrict__ OutB,
    int M, int N, int Kp, int relu)
{
  const int wave = threadIdx.x >> 5;
  const int lane = threadIdx.x & 31;
  const int half = lane >> 4;     // 0: lanes 0-15, 1: lanes 16-31
  const int l    = lane & 15;
  const int rowg = wave >> 2;     // 0..1
  const int colg = wave & 3;      // 0..3
  const int rowBase = blockIdx.y * 64 + rowg * 32;
  const int c0      = blockIdx.x * 256 + colg * 64;
  if (c0 >= N) return;            // whole-wave early out keeps EXEC full for WMMA

  const bf16_t* arow[2];
#pragma unroll
  for (int rt = 0; rt < 2; ++rt)
    arow[rt] = A + (size_t)(rowBase + rt * 16 + l) * (size_t)Kp;

  const bf16_t* wrow[4];
  int cn[4];
#pragma unroll
  for (int ct = 0; ct < 4; ++ct) {
    cn[ct] = c0 + ct * 16 + l;
    int cw = (cn[ct] < N) ? cn[ct] : (N - 1);   // clamp: OOB cols load garbage, never stored
    wrow[ct] = W + (size_t)cw * (size_t)Kp;
  }

  v8f acc[2][4];
#pragma unroll
  for (int rt = 0; rt < 2; ++rt)
#pragma unroll
    for (int ct = 0; ct < 4; ++ct)
      acc[rt][ct] = (v8f){0.f, 0.f, 0.f, 0.f, 0.f, 0.f, 0.f, 0.f};

  for (int kk = 0; kk < Kp; kk += 32) {
    v16bf af[2], bfr[4];
    // A 16x32 bf16 frag: half 0 -> K {0..7, 16..23}; half 1 -> K {8..15, 24..31}
#pragma unroll
    for (int rt = 0; rt < 2; ++rt) {
      v8bf a0 = *(const v8bf*)(arow[rt] + kk + 8 * half);
      v8bf a1 = *(const v8bf*)(arow[rt] + kk + 16 + 8 * half);
#pragma unroll
      for (int e = 0; e < 8; ++e) { af[rt][e] = a0[e]; af[rt][e + 8] = a1[e]; }
    }
    // B 32x16 bf16 frag: half 0 -> K {0..7, 16..23}; half 1 -> K {8..15, 24..31}
#pragma unroll
    for (int ct = 0; ct < 4; ++ct) {
      v8bf b0 = *(const v8bf*)(wrow[ct] + kk + 8 * half);
      v8bf b1 = *(const v8bf*)(wrow[ct] + kk + 16 + 8 * half);
#pragma unroll
      for (int e = 0; e < 8; ++e) { bfr[ct][e] = b0[e]; bfr[ct][e + 8] = b1[e]; }
    }
#pragma unroll
    for (int rt = 0; rt < 2; ++rt)
#pragma unroll
      for (int ct = 0; ct < 4; ++ct)
        acc[rt][ct] = __builtin_amdgcn_wmma_f32_16x16x32_bf16(
            false, af[rt], false, bfr[ct], (short)0, acc[rt][ct], false, false);
  }

  // Epilogue. C/D layout: acc[r] -> row (+ r + 8*half), col = tile base + l
#pragma unroll
  for (int ct = 0; ct < 4; ++ct) {
    if (cn[ct] < N) {
      float bv = bias ? bias[cn[ct]] : 0.0f;
#pragma unroll
      for (int rt = 0; rt < 2; ++rt)
#pragma unroll
        for (int r = 0; r < 8; ++r) {
          int row = rowBase + rt * 16 + r + 8 * half;
          float v = acc[rt][ct][r] + bv;
          if (relu) v = fmaxf(v, 0.0f);
          size_t o = (size_t)row * (size_t)N + (size_t)cn[ct];
          if (OutF) OutF[o] = v;
          if (OutB) OutB[o] = (bf16_t)v;
        }
    }
  }
}

// ---------------------------------------------------------------------------
// BatchNorm (training-mode batch stats over 128*8 elems) + ReLU + the torch
// reshape permutation, written as bf16 into x_feat[P][S'][128][192].
// one block per (past, channel); which = 0(board)/1(order)/2(message)
// ---------------------------------------------------------------------------
__global__ __launch_bounds__(256) void bn_relu_permute(
    const float* __restrict__ feat, const float* __restrict__ g,
    const float* __restrict__ be, bf16_t* __restrict__ xfeat, int which)
{
  __shared__ float s1[256], s2[256];
  const int p = blockIdx.x >> 6;
  const int c = blockIdx.x & 63;
  const int tid = threadIdx.x;

  float vals[4];
  float sum = 0.f, sq = 0.f;
#pragma unroll
  for (int t = 0; t < 4; ++t) {
    int i = tid + t * 256;                       // i = b*8 + s
    float v = feat[((size_t)(p * 1024 + i)) * 64 + c];
    vals[t] = v; sum += v; sq += v * v;
  }
  s1[tid] = sum; s2[tid] = sq;
  __syncthreads();
  for (int off = 128; off > 0; off >>= 1) {
    if (tid < off) { s1[tid] += s1[tid + off]; s2[tid] += s2[tid + off]; }
    __syncthreads();
  }
  float mean = s1[0] * (1.0f / 1024.0f);
  float var  = s2[0] * (1.0f / 1024.0f) - mean * mean;
  float scale = g[c] * rsqrtf(var + 1e-5f);
  float shift = be[c] - mean * scale;

  const int sp = c >> 3;       // s' = c/8
  const int c8 = c & 7;
#pragma unroll
  for (int t = 0; t < 4; ++t) {
    int i = tid + t * 256;
    int b = i >> 3, s = i & 7;
    float y = fmaxf(vals[t] * scale + shift, 0.0f);
    int bpp = (c8 << 4) + (b >> 3);              // b'' = 16*(c%8) + b/8
    int h   = 24 * (b & 7) + (which << 3) + s;   // h = 24*(b%8) + 8*which + s
    xfeat[((size_t)(p * 8 + sp) * 128 + bpp) * 192 + h] = (bf16_t)y;
  }
}

// ---------------------------------------------------------------------------
// LSTM pointwise update for step s over all pasts (rows = p*128 + b'')
// gates = grec (h@Whh^T + bhh) + Xg[(p*8+s)*128+b''] (x@Wih^T + bih)
// h is kept in bf16 (it is only ever consumed as a GEMM A operand).
// ---------------------------------------------------------------------------
__device__ __forceinline__ float sigf(float x) { return 1.0f / (1.0f + expf(-x)); }

__global__ __launch_bounds__(256) void lstm_step(
    const float* __restrict__ grec, const float* __restrict__ Xg,
    bf16_t* __restrict__ h, float* __restrict__ c,
    bf16_t* __restrict__ wf1in, int s)
{
  int idx = blockIdx.x * 256 + threadIdx.x;
  if (idx >= 1280 * 192) return;
  int row = idx / 192, j = idx - row * 192;
  int p = row >> 7, b = row & 127;
  size_t gr = (size_t)row * 768;
  size_t xr = (size_t)((p * 8 + s) * 128 + b) * 768;
  float gi = grec[gr +       j] + Xg[xr +       j];
  float gf = grec[gr + 192 + j] + Xg[xr + 192 + j];
  float gg = grec[gr + 384 + j] + Xg[xr + 384 + j];
  float go = grec[gr + 576 + j] + Xg[xr + 576 + j];
  float cn = sigf(gf) * c[idx] + sigf(gi) * tanhf(gg);
  float hn = sigf(go) * tanhf(cn);
  c[idx] = cn;
  h[idx] = (bf16_t)hn;
  wf1in[(size_t)row * 1536 + s * 192 + j] = (bf16_t)hn;  // outs laid out [b', S*H]
}

// concat relu(f1) with other_ind / me_ind -> bf16 [1280, 288] (cols 270.. zero)
__global__ __launch_bounds__(256) void build_xcat(
    const float* __restrict__ f1, const float* __restrict__ oi,
    const float* __restrict__ mi, bf16_t* __restrict__ xcat)
{
  int idx = blockIdx.x * 256 + threadIdx.x;
  if (idx >= 1280 * 288) return;
  int row = idx / 288, col = idx - row * 288;
  int p = row >> 7, b = row & 127;
  float v;
  if (col < 256)      v = f1[(size_t)row * 256 + col];
  else if (col < 263) v = oi[(size_t)(b * 10 + p) * 7 + (col - 256)];
  else if (col < 270) v = mi[(size_t)(b * 10 + p) * 7 + (col - 263)];
  else                v = 0.0f;
  xcat[(size_t)row * 288 + col] = (bf16_t)v;
}

// out[b,n] = sum_p f2[(p*128+b)*8 + n]
__global__ __launch_bounds__(256) void sum_pasts(
    const float* __restrict__ f2, float* __restrict__ out)
{
  int idx = blockIdx.x * 256 + threadIdx.x;
  if (idx >= 1024) return;
  float s = 0.f;
  for (int p = 0; p < 10; ++p) s += f2[(size_t)p * 1024 + idx];
  out[idx] = s;
}

// ---------------------------------------------------------------------------
extern "C" void kernel_launch(void* const* d_in, const int* in_sizes, int n_in,
                              void* d_out, int out_size, void* d_ws, size_t ws_size,
                              hipStream_t stream) {
  (void)in_sizes; (void)n_in; (void)out_size; (void)ws_size;
  const float* board  = (const float*)d_in[0];
  const float* order  = (const float*)d_in[1];
  const float* msg    = (const float*)d_in[2];
  const float* oi     = (const float*)d_in[3];
  const float* mi     = (const float*)d_in[4];
  const float* h0     = (const float*)d_in[5];
  const float* c0     = (const float*)d_in[6];
  const float* Wb1 = (const float*)d_in[7],  *bb1 = (const float*)d_in[8];
  const float* Wb2 = (const float*)d_in[9],  *bb2 = (const float*)d_in[10];
  const float* Wb3 = (const float*)d_in[11], *bb3 = (const float*)d_in[12];
  const float* Wo1 = (const float*)d_in[13], *bo1 = (const float*)d_in[14];
  const float* Wo2 = (const float*)d_in[15], *bo2 = (const float*)d_in[16];
  const float* Wo3 = (const float*)d_in[17], *bo3 = (const float*)d_in[18];
  const float* Wm1 = (const float*)d_in[19], *bm1 = (const float*)d_in[20];
  const float* Wm2 = (const float*)d_in[21], *bm2 = (const float*)d_in[22];
  const float* Wm3 = (const float*)d_in[23], *bm3 = (const float*)d_in[24];
  const float* g1  = (const float*)d_in[25], *be1 = (const float*)d_in[26];
  const float* g2  = (const float*)d_in[27], *be2 = (const float*)d_in[28];
  const float* g3  = (const float*)d_in[29], *be3 = (const float*)d_in[30];
  const float* Wih = (const float*)d_in[31], *Whh = (const float*)d_in[32];
  const float* bih = (const float*)d_in[33], *bhh = (const float*)d_in[34];
  const float* Wf1 = (const float*)d_in[35], *bf1 = (const float*)d_in[36];
  const float* Wf2 = (const float*)d_in[37], *bf2 = (const float*)d_in[38];

  float* ws = (float*)d_ws;
  size_t off = 0;
  auto takef = [&](size_t n) { float* p = ws + off; off += n; return p; };
  auto takeb = [&](size_t n) { bf16_t* p = (bf16_t*)(ws + off); off += n / 2; return p; };

  // bf16 staging (element counts; all even)
  bf16_t* board_b  = takeb(10240ull * 6080);  // reused later for Xg (f32, 7.86M floats < 31.1M)
  bf16_t* order_b  = takeb(10240ull * 64);
  bf16_t* msg_b    = takeb(10240ull * 32);
  bf16_t* Wb1_b = takeb(1024ull * 6080);
  bf16_t* Wb2_b = takeb(256ull * 1024);
  bf16_t* Wb3_b = takeb(64ull * 256);
  bf16_t* Wo1_b = takeb(128ull * 64);
  bf16_t* Wo2_b = takeb(32ull * 128);
  bf16_t* Wo3_b = takeb(64ull * 32);
  bf16_t* Wm1_b = takeb(64ull * 32);
  bf16_t* Wm2_b = takeb(32ull * 64);
  bf16_t* Wm3_b = takeb(64ull * 32);
  bf16_t* Wih_b = takeb(768ull * 192);
  bf16_t* Whh_b = takeb(768ull * 192);
  bf16_t* Wf1_b = takeb(256ull * 1536);
  bf16_t* Wf2_b = takeb(8ull * 288);
  // bf16 activations (only ever consumed by the next GEMM)
  bf16_t* bfeat1 = takeb(10240ull * 1024);
  bf16_t* bfeat2 = takeb(10240ull * 256);
  bf16_t* ofeat1 = takeb(10240ull * 128);
  bf16_t* ofeat2 = takeb(10240ull * 32);
  bf16_t* mfeat1 = takeb(10240ull * 64);
  bf16_t* mfeat2 = takeb(10240ull * 32);
  bf16_t* xfeat  = takeb(10240ull * 192);
  bf16_t* hbuf   = takeb(1280ull * 192);
  bf16_t* wf1in  = takeb(1280ull * 1536);
  bf16_t* xcat   = takeb(1280ull * 288);
  // f32 buffers (consumed by pointwise math)
  float* bfeat3 = takef(655360);
  float* ofeat3 = takef(655360);
  float* mfeat3 = takef(655360);
  float* grec   = takef(983040);
  float* cbuf   = takef(245760);
  float* f1out  = takef(327680);
  float* f2out  = takef(10240);
  float* Xg = (float*)board_b;   // board_b dead after board L1; Xg fits inside

  const dim3 blk(256);
  auto g1d = [](size_t n) { return dim3((unsigned)((n + 255) / 256)); };

  // ---- stage everything the WMMAs touch as padded bf16 ----
  cvt_perm_bf16<<<g1d(10240ull * 6080), blk, 0, stream>>>(board, board_b, 6075, 6080);
  cvt_perm_bf16<<<g1d(10240ull * 64),   blk, 0, stream>>>(order, order_b, 40, 64);
  cvt_perm_bf16<<<g1d(10240ull * 32),   blk, 0, stream>>>(msg,   msg_b,   20, 32);
  cvt_pad_bf16<<<g1d(1024ull * 6080), blk, 0, stream>>>(Wb1, Wb1_b, 1024, 6075, 6080);
  cvt_pad_bf16<<<g1d(256ull * 1024),  blk, 0, stream>>>(Wb2, Wb2_b, 256, 1024, 1024);
  cvt_pad_bf16<<<g1d(64ull * 256),    blk, 0, stream>>>(Wb3, Wb3_b, 64, 256, 256);
  cvt_pad_bf16<<<g1d(128ull * 64),    blk, 0, stream>>>(Wo1, Wo1_b, 128, 40, 64);
  cvt_pad_bf16<<<g1d(32ull * 128),    blk, 0, stream>>>(Wo2, Wo2_b, 32, 128, 128);
  cvt_pad_bf16<<<g1d(64ull * 32),     blk, 0, stream>>>(Wo3, Wo3_b, 64, 32, 32);
  cvt_pad_bf16<<<g1d(64ull * 32),     blk, 0, stream>>>(Wm1, Wm1_b, 64, 20, 32);
  cvt_pad_bf16<<<g1d(32ull * 64),     blk, 0, stream>>>(Wm2, Wm2_b, 32, 64, 64);
  cvt_pad_bf16<<<g1d(64ull * 32),     blk, 0, stream>>>(Wm3, Wm3_b, 64, 32, 32);
  cvt_pad_bf16<<<g1d(768ull * 192),   blk, 0, stream>>>(Wih, Wih_b, 768, 192, 192);
  cvt_pad_bf16<<<g1d(768ull * 192),   blk, 0, stream>>>(Whh, Whh_b, 768, 192, 192);
  cvt_pad_bf16<<<g1d(256ull * 1536),  blk, 0, stream>>>(Wf1, Wf1_b, 256, 1536, 1536);
  cvt_pad_bf16<<<g1d(8ull * 288),     blk, 0, stream>>>(Wf2, Wf2_b, 8, 270, 288);
  cvt_pad_bf16<<<g1d(1280ull * 192),  blk, 0, stream>>>(h0, hbuf, 1280, 192, 192);

  // ---- board MLP: 6075 -> 1024 -> 256 -> 64   (rows 10240 -> grid.y = 160)
  gemm_wmma<<<dim3(4, 160), blk, 0, stream>>>(board_b, Wb1_b, bb1, nullptr, bfeat1, 10240, 1024, 6080, 1);
  gemm_wmma<<<dim3(1, 160), blk, 0, stream>>>(bfeat1,  Wb2_b, bb2, nullptr, bfeat2, 10240,  256, 1024, 1);
  gemm_wmma<<<dim3(1, 160), blk, 0, stream>>>(bfeat2,  Wb3_b, bb3, bfeat3, nullptr, 10240,   64,  256, 0);
  // ---- order MLP: 40 -> 128 -> 32 -> 64
  gemm_wmma<<<dim3(1, 160), blk, 0, stream>>>(order_b, Wo1_b, bo1, nullptr, ofeat1, 10240,  128,   64, 1);
  gemm_wmma<<<dim3(1, 160), blk, 0, stream>>>(ofeat1,  Wo2_b, bo2, nullptr, ofeat2, 10240,   32,  128, 1);
  gemm_wmma<<<dim3(1, 160), blk, 0, stream>>>(ofeat2,  Wo3_b, bo3, ofeat3, nullptr, 10240,   64,   32, 0);
  // ---- message MLP: 20 -> 64 -> 32 -> 64
  gemm_wmma<<<dim3(1, 160), blk, 0, stream>>>(msg_b,   Wm1_b, bm1, nullptr, mfeat1, 10240,   64,   32, 1);
  gemm_wmma<<<dim3(1, 160), blk, 0, stream>>>(mfeat1,  Wm2_b, bm2, nullptr, mfeat2, 10240,   32,   64, 1);
  gemm_wmma<<<dim3(1, 160), blk, 0, stream>>>(mfeat2,  Wm3_b, bm3, mfeat3, nullptr, 10240,   64,   32, 0);
  // ---- BN + ReLU + reshape permutation into x_feat (bf16)
  bn_relu_permute<<<640, blk, 0, stream>>>(bfeat3, g1, be1, xfeat, 0);
  bn_relu_permute<<<640, blk, 0, stream>>>(ofeat3, g2, be2, xfeat, 1);
  bn_relu_permute<<<640, blk, 0, stream>>>(mfeat3, g3, be3, xfeat, 2);
  // ---- Xg = x_feat @ Wih^T + bih  (all pasts & steps at once; overwrites board_b)
  gemm_wmma<<<dim3(3, 160), blk, 0, stream>>>(xfeat, Wih_b, bih, Xg, nullptr, 10240, 768, 192, 0);
  // ---- LSTM state init (h staged bf16 above; c stays f32)
  hipMemcpyAsync(cbuf, c0, (size_t)1280 * 192 * sizeof(float), hipMemcpyDeviceToDevice, stream);
  // ---- 8 recurrent steps, all 10 pasts batched (M=1280 -> grid.y = 20)
  for (int s = 0; s < 8; ++s) {
    gemm_wmma<<<dim3(3, 20), blk, 0, stream>>>(hbuf, Whh_b, bhh, grec, nullptr, 1280, 768, 192, 0);
    lstm_step<<<960, blk, 0, stream>>>(grec, Xg, hbuf, cbuf, wf1in, s);
  }
  // ---- final head
  gemm_wmma<<<dim3(1, 20), blk, 0, stream>>>(wf1in, Wf1_b, bf1, f1out, nullptr, 1280, 256, 1536, 1);
  build_xcat<<<1440, blk, 0, stream>>>(f1out, oi, mi, xcat);
  gemm_wmma<<<dim3(1, 20), blk, 0, stream>>>(xcat, Wf2_b, bf2, f2out, nullptr, 1280, 8, 288, 0);
  sum_pasts<<<4, blk, 0, stream>>>(f2out, (float*)d_out);
}
